// MultiHeadAttention_50465865728115
// MI455X (gfx1250) — matrix operand order repaired and checked by reading, not compiled, not run on hardware
//
#include <hip/hip_runtime.h>

// ---------------------------------------------------------------------------
// MHA forward for MI455X (gfx1250, wave32, WMMA 16x16x32 f16 -> f32 acc,
// TDM async tensor loads for tile staging)
// ---------------------------------------------------------------------------

#define D_MODEL 1024
#define N_HEAD  16
#define D_K     64
#define SEQ     2048
#define BATCH   4

typedef __attribute__((ext_vector_type(16))) _Float16     v16h;
typedef __attribute__((ext_vector_type(8)))  _Float16     v8h;
typedef __attribute__((ext_vector_type(4)))  _Float16     h4_t;
typedef __attribute__((ext_vector_type(8)))  float        v8f;
typedef __attribute__((ext_vector_type(4)))  unsigned int u32x4;
typedef __attribute__((ext_vector_type(8)))  int          i32x8;
typedef __attribute__((ext_vector_type(4)))  int          i32x4;

#if defined(__has_builtin)
#if __has_builtin(__builtin_amdgcn_tensor_load_to_lds) && \
    __has_builtin(__builtin_amdgcn_s_wait_tensorcnt)
#define USE_TDM 1
#endif
#endif
#ifndef USE_TDM
#define USE_TDM 0
#endif

__device__ __forceinline__ v16h cat8(v8h lo, v8h hi) {
    return __builtin_shufflevector(lo, hi, 0,1,2,3,4,5,6,7,8,9,10,11,12,13,14,15);
}

__device__ __forceinline__ v8f wmma_f16(v16h a, v16h b, v8f c) {
    return __builtin_amdgcn_wmma_f32_16x16x32_f16(
        /*neg_a=*/false, a, /*neg_b=*/false, b,
        /*c_mod=*/(short)0, c, /*reuse_a=*/false, /*reuse_b=*/false);
}

#if USE_TDM
// ---------------------------------------------------------------------------
// Issue one TDM 2D tile load: tile_d0 x tile_d1 elements of 2 bytes, row
// stride `stride_elems`, global tile start `gaddr`, LDS byte offset `lds_off`.
// D# layout per CDNA5 ISA (groups 2/3 zero => 2D).
// 6-arg builtin form (clang-23 / therock lane).
// ---------------------------------------------------------------------------
__device__ __forceinline__ void tdm_load_2d(unsigned int lds_off, const void* gaddr,
                                            unsigned int tile_d0, unsigned int tile_d1,
                                            unsigned long long stride_elems)
{
    const unsigned long long ga = (unsigned long long)(uintptr_t)gaddr;
    const unsigned int td0 = 1u << 30;   // huge tensor dims: no OOB clipping
    const unsigned int td1 = 1u << 30;
    u32x4 g0;
    g0[0] = 1u;                                            // count=1, user mode
    g0[1] = lds_off;                                       // lds_addr (bytes)
    g0[2] = (unsigned int)ga;                              // global_addr[31:0]
    g0[3] = (unsigned int)((ga >> 32) & 0x01FFFFFFu) | 0x80000000u; // addr[56:32] | type=2
    i32x8 g1;
    g1[0] = (int)(1u << 16);                               // data_size=1 (2 bytes)
    g1[1] = (int)((td0 & 0xFFFFu) << 16);                  // tensor_dim0 lo
    g1[2] = (int)((td0 >> 16) | ((td1 & 0xFFFFu) << 16));  // dim0 hi | dim1 lo
    g1[3] = (int)((td1 >> 16) | (tile_d0 << 16));          // dim1 hi | tile_dim0
    g1[4] = (int)(tile_d1 & 0xFFFFu);                      // tile_dim1 (tile_dim2=0)
    g1[5] = (int)(unsigned int)(stride_elems);             // dim0 stride lo
    g1[6] = (int)(unsigned int)((stride_elems >> 32) & 0xFFFFull);
    g1[7] = 0;
    const i32x4 z4 = {0, 0, 0, 0};
    const i32x8 z8 = {0, 0, 0, 0, 0, 0, 0, 0};
    __builtin_amdgcn_tensor_load_to_lds(g0, g1, z4, z4, z8, 0);
}
#endif

// ---------------------------------------------------------------------------
// Elementwise f32 -> f16 convert (4 elems/thread, float4 in / h4 out)
// ---------------------------------------------------------------------------
__global__ __launch_bounds__(256)
void cvt_f32_to_f16_kernel(const float* __restrict__ src, _Float16* __restrict__ dst)
{
    const size_t i = ((size_t)blockIdx.x * blockDim.x + threadIdx.x) * 4;
    float4 f = *(const float4*)(src + i);
    h4_t h = { (_Float16)f.x, (_Float16)f.y, (_Float16)f.z, (_Float16)f.w };
    *(h4_t*)(dst + i) = h;
}

// ---------------------------------------------------------------------------
// GEMM: Out[M,N] = A[M,K] * W[N,K]^T + bias[N]   (A, W are f16)
// Block: 256 threads (8 waves). Tile 128(M) x 128(N), K-chunks of 32,
// double-buffered LDS filled by TDM (waves 0/1 drive the DMA).
// ---------------------------------------------------------------------------
template<bool OUT_F16>
__global__ __launch_bounds__(256)
void gemm_wmma_kernel(const _Float16* __restrict__ A,
                      const _Float16* __restrict__ W,
                      const float* __restrict__ bias,
                      void* __restrict__ Outp,
                      int M, int N, int K)
{
    __shared__ _Float16 ldsA[2][128][32];   // [m][k], k contiguous
    __shared__ _Float16 ldsB[2][128][32];   // [n][k], k contiguous

    const int t    = threadIdx.x;
    const int lane = t & 31;
    const int w    = t >> 5;
    const int g    = lane >> 4;
    const int ln   = lane & 15;
    const int wm   = w & 1;
    const int wn   = w >> 1;
    const int tM   = blockIdx.x * 128;
    const int tN   = blockIdx.y * 128;

    const v8f vzero = {0.f,0.f,0.f,0.f,0.f,0.f,0.f,0.f};
    v8f acc[4][2];
    #pragma unroll
    for (int mt = 0; mt < 4; ++mt)
        #pragma unroll
        for (int nt = 0; nt < 2; ++nt) acc[mt][nt] = vzero;

    const int NC = K / 32;

#if USE_TDM
    // prologue: chunk 0 -> buffer 0
    if (w == 0) tdm_load_2d((unsigned int)(uintptr_t)&ldsA[0][0][0],
                            A + (size_t)tM * K, 32, 128, (unsigned long long)K);
    if (w == 1) tdm_load_2d((unsigned int)(uintptr_t)&ldsB[0][0][0],
                            W + (size_t)tN * K, 32, 128, (unsigned long long)K);
#endif

    for (int c = 0; c < NC; ++c) {
        const int buf = c & 1;
#if USE_TDM
        if (w < 2) __builtin_amdgcn_s_wait_tensorcnt(0);
        __syncthreads();                       // chunk c resident; chunk c-1 consumed
        if (c + 1 < NC) {
            const int nb = (c + 1) & 1;
            const int k1 = (c + 1) * 32;
            if (w == 0) tdm_load_2d((unsigned int)(uintptr_t)&ldsA[nb][0][0],
                                    A + (size_t)tM * K + k1, 32, 128, (unsigned long long)K);
            if (w == 1) tdm_load_2d((unsigned int)(uintptr_t)&ldsB[nb][0][0],
                                    W + (size_t)tN * K + k1, 32, 128, (unsigned long long)K);
        }
#else
        const int k0 = c * 32;
        __syncthreads();
        {   // manual staging fallback: 2 threads per row, v8h pairs
            const int r = t >> 1, hh = t & 1;
            const _Float16* srcA = A + (size_t)(tM + r) * K + k0 + hh * 16;
            *(v8h*)&ldsA[buf][r][hh*16]     = *(const v8h*)(srcA);
            *(v8h*)&ldsA[buf][r][hh*16 + 8] = *(const v8h*)(srcA + 8);
            const _Float16* srcW = W + (size_t)(tN + r) * K + k0 + hh * 16;
            *(v8h*)&ldsB[buf][r][hh*16]     = *(const v8h*)(srcW);
            *(v8h*)&ldsB[buf][r][hh*16 + 8] = *(const v8h*)(srcW + 8);
        }
        __syncthreads();
#endif
        // ---- fragments (A: 16x32 layout, B: 32x16 layout per ISA tables) ----
        v16h aF[4];
        #pragma unroll
        for (int mt = 0; mt < 4; ++mt) {
            const int row = wm * 64 + mt * 16 + ln;
            v8h lo = *(const v8h*)&ldsA[buf][row][8*g];
            v8h hi = *(const v8h*)&ldsA[buf][row][16 + 8*g];
            aF[mt] = cat8(lo, hi);
        }
        v16h bF[2];
        #pragma unroll
        for (int nt = 0; nt < 2; ++nt) {
            const int n = wn * 32 + nt * 16 + ln;
            v8h lo = *(const v8h*)&ldsB[buf][n][8*g];
            v8h hi = *(const v8h*)&ldsB[buf][n][16 + 8*g];
            bF[nt] = cat8(lo, hi);
        }
        #pragma unroll
        for (int mt = 0; mt < 4; ++mt)
            #pragma unroll
            for (int nt = 0; nt < 2; ++nt)
                acc[mt][nt] = wmma_f16(aF[mt], bF[nt], acc[mt][nt]);
    }

    // ---- epilogue: bias + store (C layout: vgpr r -> row r or r+8) ----
    #pragma unroll
    for (int nt = 0; nt < 2; ++nt) {
        const int n = tN + wn * 32 + nt * 16 + ln;
        const float bv = bias[n];
        #pragma unroll
        for (int mt = 0; mt < 4; ++mt) {
            #pragma unroll
            for (int r = 0; r < 8; ++r) {
                const int mrow = tM + wm * 64 + mt * 16 + r + 8 * g;
                const float val = acc[mt][nt][r] + bv;
                if (OUT_F16)
                    ((_Float16*)Outp)[(size_t)mrow * N + n] = (_Float16)val;
                else
                    ((float*)Outp)[(size_t)mrow * N + n] = val;
            }
        }
    }
}

// ---------------------------------------------------------------------------
// Flash attention: one block per (64-query tile, head, batch). 128 threads.
// Q tile + K chunks staged by TDM; V transposed via VGPR staging.
// ---------------------------------------------------------------------------
__global__ __launch_bounds__(128)
void attn_kernel(const _Float16* __restrict__ qp,
                 const _Float16* __restrict__ kp,
                 const _Float16* __restrict__ vp,
                 _Float16* __restrict__ ao)
{
    __shared__ _Float16 ldsQ[64][64];   // [q][d]
    __shared__ _Float16 ldsK[32][64];   // [kk][d]
    __shared__ _Float16 ldsV[64][32];   // [d][kk] (transposed)
    __shared__ _Float16 ldsP[64][32];   // per-wave P scratch

    const int t    = threadIdx.x;
    const int lane = t & 31;
    const int w    = t >> 5;
    const int g    = lane >> 4;
    const int ln   = lane & 15;
    const int qT   = blockIdx.x;
    const int h    = blockIdx.y;
    const int b    = blockIdx.z;

    const size_t headOff = (size_t)b * SEQ * D_MODEL + (size_t)h * D_K;
    const _Float16* qbase = qp + headOff + (size_t)qT * 64 * D_MODEL;
    const _Float16* kbase = kp + headOff;
    const _Float16* vbase = vp + headOff;

    // ---- Q tile (64 x 64) ----
#if USE_TDM
    if (w == 0) {
        tdm_load_2d((unsigned int)(uintptr_t)&ldsQ[0][0], qbase, 64, 64,
                    (unsigned long long)D_MODEL);
        __builtin_amdgcn_s_wait_tensorcnt(0);
    }
    __syncthreads();
#else
    {
        const int r = t >> 1, hh = t & 1;
        const _Float16* src = qbase + (size_t)r * D_MODEL + hh * 32;
        #pragma unroll
        for (int i = 0; i < 4; ++i)
            *(v8h*)&ldsQ[r][hh*32 + i*8] = *(const v8h*)(src + i*8);
    }
    __syncthreads();
#endif

    // ---- Q A-fragments (2 d-chunks of 32), loop-invariant ----
    v16h aQ[2];
    {
        const int row = w * 16 + ln;
        #pragma unroll
        for (int dc = 0; dc < 2; ++dc) {
            v8h lo = *(const v8h*)&ldsQ[row][dc*32 + 8*g];
            v8h hi = *(const v8h*)&ldsQ[row][dc*32 + 16 + 8*g];
            aQ[dc] = cat8(lo, hi);
        }
    }

    const v8f vzero = {0.f,0.f,0.f,0.f,0.f,0.f,0.f,0.f};
    v8f acc[4];
    #pragma unroll
    for (int nt = 0; nt < 4; ++nt) acc[nt] = vzero;
    float m_st[8], l_st[8];
    #pragma unroll
    for (int r = 0; r < 8; ++r) { m_st[r] = -1e30f; l_st[r] = 0.f; }

    const float scale = 0.125f;   // 1/sqrt(64)

    for (int j = 0; j < SEQ / 32; ++j) {
        __syncthreads();   // previous chunk fully consumed
        // ---- stage K chunk: 32 rows x 64 halves ----
#if USE_TDM
        if (w == 0)
            tdm_load_2d((unsigned int)(uintptr_t)&ldsK[0][0],
                        kbase + (size_t)(j * 32) * D_MODEL, 64, 32,
                        (unsigned long long)D_MODEL);
#else
        {
            const int r = t >> 2, i = t & 3;
            const _Float16* src = kbase + (size_t)(j * 32 + r) * D_MODEL;
            *(v8h*)&ldsK[r][i*8]      = *(const v8h*)(src + i*8);
            *(v8h*)&ldsK[r][32 + i*8] = *(const v8h*)(src + 32 + i*8);
        }
#endif
        // ---- stage V chunk transposed: ldsV[d][kk] ----
        {
            const int r = t >> 2, i = t & 3;
            const _Float16* src = vbase + (size_t)(j * 32 + r) * D_MODEL + i * 16;
            v8h a0 = *(const v8h*)(src);
            v8h a1 = *(const v8h*)(src + 8);
            #pragma unroll
            for (int e = 0; e < 8; ++e) {
                ldsV[i*16 + e][r]     = a0[e];
                ldsV[i*16 + 8 + e][r] = a1[e];
            }
        }
#if USE_TDM
        if (w == 0) __builtin_amdgcn_s_wait_tensorcnt(0);
#endif
        __syncthreads();

        // ---- scores: S = Q * K^T (two 16x16 tiles over kk) ----
        v8f s0 = vzero, s1 = vzero;
        #pragma unroll
        for (int dc = 0; dc < 2; ++dc) {
            {
                v8h lo = *(const v8h*)&ldsK[ln][dc*32 + 8*g];
                v8h hi = *(const v8h*)&ldsK[ln][dc*32 + 16 + 8*g];
                s0 = wmma_f16(aQ[dc], cat8(lo, hi), s0);
            }
            {
                v8h lo = *(const v8h*)&ldsK[16 + ln][dc*32 + 8*g];
                v8h hi = *(const v8h*)&ldsK[16 + ln][dc*32 + 16 + 8*g];
                s1 = wmma_f16(aQ[dc], cat8(lo, hi), s1);
            }
        }

        // ---- online softmax (row stats per (vgpr r, lane-half g)) ----
        #pragma unroll
        for (int r = 0; r < 8; ++r) {
            float v = fmaxf(s0[r], s1[r]);
            v = fmaxf(v, __shfl_xor(v, 1));
            v = fmaxf(v, __shfl_xor(v, 2));
            v = fmaxf(v, __shfl_xor(v, 4));
            v = fmaxf(v, __shfl_xor(v, 8));
            const float mN = fmaxf(m_st[r], v * scale);
            const float f  = __expf(m_st[r] - mN);
            const float e0 = __expf(s0[r] * scale - mN);
            const float e1 = __expf(s1[r] * scale - mN);
            float rs = e0 + e1;
            rs += __shfl_xor(rs, 1);
            rs += __shfl_xor(rs, 2);
            rs += __shfl_xor(rs, 4);
            rs += __shfl_xor(rs, 8);
            l_st[r] = l_st[r] * f + rs;
            m_st[r] = mN;
            #pragma unroll
            for (int nt = 0; nt < 4; ++nt) acc[nt][r] *= f;
            const int prow = w * 16 + r + 8 * g;
            ldsP[prow][ln]      = (_Float16)e0;
            ldsP[prow][16 + ln] = (_Float16)e1;
        }

        // ---- PV: acc += P(16x32) * V(32x64) ----
        {
            const int row = w * 16 + ln;
            v8h lo = *(const v8h*)&ldsP[row][8*g];
            v8h hi = *(const v8h*)&ldsP[row][16 + 8*g];
            const v16h aP = cat8(lo, hi);
            #pragma unroll
            for (int nt = 0; nt < 4; ++nt) {
                const int d = nt * 16 + ln;
                v8h blo = *(const v8h*)&ldsV[d][8*g];
                v8h bhi = *(const v8h*)&ldsV[d][16 + 8*g];
                acc[nt] = wmma_f16(aP, cat8(blo, bhi), acc[nt]);
            }
        }
    }

    // ---- normalize and store merged-head f16 output ----
    _Float16* outp = ao + headOff + (size_t)qT * 64 * D_MODEL;
    #pragma unroll
    for (int nt = 0; nt < 4; ++nt) {
        const int d = nt * 16 + ln;
        #pragma unroll
        for (int r = 0; r < 8; ++r) {
            const int row = w * 16 + r + 8 * g;
            const float ov = acc[nt][r] / l_st[r];
            outp[(size_t)row * D_MODEL + d] = (_Float16)ov;
        }
    }
}

// ---------------------------------------------------------------------------
// Host-side launch
// ---------------------------------------------------------------------------
extern "C" void kernel_launch(void* const* d_in, const int* in_sizes, int n_in,
                              void* d_out, int out_size, void* d_ws, size_t ws_size,
                              hipStream_t stream) {
    (void)in_sizes; (void)n_in; (void)out_size; (void)ws_size;

    const float* q   = (const float*)d_in[0];
    const float* k   = (const float*)d_in[1];
    const float* v   = (const float*)d_in[2];
    const float* w_q = (const float*)d_in[3];
    const float* b_q = (const float*)d_in[4];
    const float* w_k = (const float*)d_in[5];
    const float* b_k = (const float*)d_in[6];
    const float* w_v = (const float*)d_in[7];
    const float* b_v = (const float*)d_in[8];
    const float* w_o = (const float*)d_in[9];
    const float* b_o = (const float*)d_in[10];

    const int M = BATCH * SEQ;                    // 8192
    const size_t xe = (size_t)M * D_MODEL;        // 8Mi elements
    const size_t we = (size_t)D_MODEL * D_MODEL;  // 1Mi elements

    _Float16* p = (_Float16*)d_ws;
    _Float16* qh  = p;             p += xe;       // f16 copies of inputs
    _Float16* kh  = p;             p += xe;
    _Float16* vh  = p;             p += xe;
    _Float16* wqh = p;             p += we;       // f16 copies of weights
    _Float16* wkh = p;             p += we;
    _Float16* wvh = p;             p += we;
    _Float16* woh = p;             p += we;
    _Float16* qp  = p;             p += xe;       // projected Q/K/V (f16)
    _Float16* kp  = p;             p += xe;
    _Float16* vp  = p;             p += xe;
    _Float16* ao  = p;                            // merged-head attn out (f16)

    // f32 -> f16 conversions
    cvt_f32_to_f16_kernel<<<dim3(xe / 4 / 256), dim3(256), 0, stream>>>(q, qh);
    cvt_f32_to_f16_kernel<<<dim3(xe / 4 / 256), dim3(256), 0, stream>>>(k, kh);
    cvt_f32_to_f16_kernel<<<dim3(xe / 4 / 256), dim3(256), 0, stream>>>(v, vh);
    cvt_f32_to_f16_kernel<<<dim3(we / 4 / 256), dim3(256), 0, stream>>>(w_q, wqh);
    cvt_f32_to_f16_kernel<<<dim3(we / 4 / 256), dim3(256), 0, stream>>>(w_k, wkh);
    cvt_f32_to_f16_kernel<<<dim3(we / 4 / 256), dim3(256), 0, stream>>>(w_v, wvh);
    cvt_f32_to_f16_kernel<<<dim3(we / 4 / 256), dim3(256), 0, stream>>>(w_o, woh);

    dim3 gblk(256);
    dim3 ggrid(M / 128, D_MODEL / 128);           // 64 x 8

    // Q/K/V projections: f16 in, f16 out
    gemm_wmma_kernel<true><<<ggrid, gblk, 0, stream>>>(qh, wqh, b_q, qp, M, D_MODEL, D_MODEL);
    gemm_wmma_kernel<true><<<ggrid, gblk, 0, stream>>>(kh, wkh, b_k, kp, M, D_MODEL, D_MODEL);
    gemm_wmma_kernel<true><<<ggrid, gblk, 0, stream>>>(vh, wvh, b_v, vp, M, D_MODEL, D_MODEL);

    // Attention
    attn_kernel<<<dim3(SEQ / 64, N_HEAD, BATCH), dim3(128), 0, stream>>>(qp, kp, vp, ao);

    // Output projection: f16 in, f32 out
    gemm_wmma_kernel<false><<<ggrid, gblk, 0, stream>>>(ao, woh, b_o, d_out, M, D_MODEL, D_MODEL);
}
